// CorrBlock_43267500540365
// MI455X (gfx1250) — matrix operand order repaired and checked by reading, not compiled, not run on hardware
//
#include <hip/hip_runtime.h>

typedef __bf16 bf16_t;
typedef __attribute__((ext_vector_type(16))) __bf16 v16bf;
typedef __attribute__((ext_vector_type(8)))  __bf16 v8bf;
typedef __attribute__((ext_vector_type(8)))  float  v8f;

#define B_   2
#define C_   256
#define H_   96
#define W_   96
#define HW_  (H_*W_)
#define NPIX (B_*HW_)        // 18432
#define NCH  324             // 4 levels * 9*9
#define NTILE 7              // ceil(100/16)

// workspace layout, offsets in bf16 elements (total ~21 MB)
#define OFF_F1P  0u
#define OFF_L0   4718592u
#define OFF_L1   9437184u
#define OFF_L2   10616832u
#define OFF_L3   10911744u
#define OFF_ZERO 10985472u
#define WS_ELEMS 10985728u   // divisible by 256

// ---------------------------------------------------------------------------
// Kernel 1: pack f1 channel-last bf16; build pooled-f2 pyramid channel-last
// bf16 (pooling in fp32 straight from the source level for accuracy); write a
// 256-entry zero column used for zero-padding out-of-bounds samples.
// ---------------------------------------------------------------------------
__global__ __launch_bounds__(256) void corr_prep(const float* __restrict__ f1,
                                                 const float* __restrict__ f2,
                                                 bf16_t* __restrict__ ws) {
  unsigned e = blockIdx.x * 256u + threadIdx.x;
  if (e < OFF_L0) {                       // pack f1: ws[b, y, x, c] = f1[b, c, y, x]
    unsigned b = e / (HW_*C_);
    unsigned r = e % (HW_*C_);
    unsigned pos = r / C_, c = r % C_;
    unsigned y = pos / W_, x = pos % W_;
    ws[OFF_F1P + e] = (bf16_t)f1[((b*C_ + c)*H_ + y)*W_ + x];
    return;
  }
  if (e >= WS_ELEMS) return;
  if (e >= OFF_ZERO) { ws[e] = (bf16_t)0.0f; return; }

  int lvl; unsigned idx;
  if      (e < OFF_L1) { lvl = 0; idx = e - OFF_L0; }
  else if (e < OFF_L2) { lvl = 1; idx = e - OFF_L1; }
  else if (e < OFF_L3) { lvl = 2; idx = e - OFF_L2; }
  else                 { lvl = 3; idx = e - OFF_L3; }

  int Wl = W_ >> lvl, s = 1 << lvl;
  unsigned per_b = (unsigned)(Wl*Wl*C_);
  unsigned b = idx / per_b;
  unsigned r = idx % per_b;
  unsigned pos = r / C_, c = r % C_;
  unsigned yl = pos / (unsigned)Wl, xl = pos % (unsigned)Wl;
  const float* src = f2 + ((size_t)(b*C_ + c)*H_ + yl*s)*W_ + xl*s;
  float acc = 0.0f;
  for (int dy = 0; dy < s; ++dy)
    for (int dx = 0; dx < s; ++dx)
      acc += src[dy*W_ + dx];
  ws[e] = (bf16_t)(acc * (1.0f / (float)(s*s)));
}

// ---------------------------------------------------------------------------
// Kernel 2: one wave handles 4 pixels. Per (pixel, level): gather 10x10 patch
// of pyramid columns (16 per WMMA N-tile); stage all 8 B fragments (16x
// global_load_b128, kept ahead of the math by a sched_barrier so the wave has
// 16 loads in flight), then run the 8-deep v_wmma_f32_16x16x32_bf16
// accumulate chain; row -> LDS; bilinear combine + scatter.
// ---------------------------------------------------------------------------
__global__ __launch_bounds__(256) void corr_main(const float* __restrict__ cent,
                                                 const bf16_t* __restrict__ ws,
                                                 float* __restrict__ out) {
  __shared__ float Dbuf[8][112];

  const unsigned tid  = threadIdx.x;
  const unsigned wave = tid >> 5;
  const unsigned lane = tid & 31u;
  const unsigned wavebase = (blockIdx.x * 8u + wave) * 4u;   // first of 4 pixels

  const bf16_t* f1p = ws + OFF_F1P;

  // ---- preload A (16x32 bf16 fragments for all 8 K-steps of C=256) ----
  // A row m = f1 of pixel (m % 4). k order used here (same as B): lanes 0-15 hold row m with
  // K = 0..15 +32k ; lanes 16-31 hold K = 16..31 +32k.
  const unsigned mypix = wavebase + (lane & 3u);
  const bf16_t* acol = f1p + (size_t)mypix * C_;
  const int aoff = (lane < 16u) ? 0 : 16;
  v16bf A[8];
#pragma unroll
  for (int k = 0; k < 8; ++k) {
    v8bf lo = *(const v8bf*)(acol + k*32 + aoff);
    v8bf hi = *(const v8bf*)(acol + k*32 + aoff + 8);
    v16bf a;
#pragma unroll
    for (int j = 0; j < 8; ++j) { a[j] = lo[j]; a[j+8] = hi[j]; }
    A[k] = a;
  }

  const unsigned boff = (lane < 16u) ? 0u : 16u;  // B: lanes 0-15 K=0..15, 16-31 K=16..31

#pragma unroll 1
  for (int p = 0; p < 4; ++p) {
    const unsigned gp = wavebase + (unsigned)p;
    const unsigned b  = gp / HW_;
    const unsigned rr = gp % HW_;
    const unsigned y  = rr / W_, x = rr % W_;
    const float cx = cent[((size_t)(b*2u + 0u)*H_ + y)*W_ + x];
    const float cy = cent[((size_t)(b*2u + 1u)*H_ + y)*W_ + x];

#pragma unroll 1
    for (int lvl = 0; lvl < 4; ++lvl) {
      const float sc  = 1.0f / (float)(1 << lvl);
      const float clx = cx * sc, cly = cy * sc;
      const float fx = floorf(clx), fy = floorf(cly);
      const float wx1 = clx - fx, wy1 = cly - fy;
      const float wx0 = 1.0f - wx1, wy0 = 1.0f - wy1;
      const int xb = (int)fx - 4, yb = (int)fy - 4;
      const int Wl = W_ >> lvl;
      const unsigned loff = (lvl == 0) ? OFF_L0 : (lvl == 1) ? OFF_L1
                          : (lvl == 2) ? OFF_L2 : OFF_L3;
      // 32-bit element offset of this (batch, level) slab inside ws
      const unsigned slab = loff + b * (unsigned)(Wl*Wl) * C_;

#pragma unroll 1
      for (int t = 0; t < NTILE; ++t) {
        const int pos = t*16 + (int)(lane & 15u);     // patch index px*10+py
        const int px = pos / 10, py = pos % 10;
        const int gx = xb + px, gy = yb + py;
        const bool ok = (pos < 100) & (gx >= 0) & (gx < Wl) & (gy >= 0) & (gy < Wl);
        // branchless: select a 32-bit element offset, add to ws
        const unsigned colofs = ok ? (slab + (unsigned)(gy*Wl + gx) * C_ + boff)
                                   : (OFF_ZERO + boff);
        const bf16_t* col = ws + colofs;

        // stage all 8 B fragments first: 16 loads in flight
        v16bf Bv[8];
#pragma unroll
        for (int k = 0; k < 8; ++k)
          Bv[k] = *(const v16bf*)(col + (k << 5));

        // keep all loads issued before any WMMA consumes them -> progressive
        // s_wait_loadcnt instead of wait-0 per K-step
        __builtin_amdgcn_sched_barrier(0);

        v8f acc = {0.f,0.f,0.f,0.f,0.f,0.f,0.f,0.f};
#pragma unroll
        for (int k = 0; k < 8; ++k)
          acc = __builtin_amdgcn_wmma_f32_16x16x32_bf16(
              false, A[k], false, Bv[k], (short)0, acc, false, false);

        // extract row p (rows 0..3 hold pixels 0..3; lanes 0-15, N = lane)
        const float v = (p == 0) ? acc[0] : (p == 1) ? acc[1]
                      : (p == 2) ? acc[2] : acc[3];
        if ((lane < 16u) && (pos < 100)) Dbuf[wave][pos] = v;
      }
      __syncthreads();

      // bilinear combine: out channel o = i*9 + j, i = x-offset, j = y-offset
#pragma unroll 1
      for (int rep = 0; rep < 3; ++rep) {
        const int o = (int)lane + rep*32;
        if (o < 81) {
          const int i = o / 9, j = o % 9;
          const float d00 = Dbuf[wave][i*10 + j];
          const float d10 = Dbuf[wave][(i+1)*10 + j];
          const float d01 = Dbuf[wave][i*10 + j + 1];
          const float d11 = Dbuf[wave][(i+1)*10 + j + 1];
          const float val = (wx0*(wy0*d00 + wy1*d01) + wx1*(wy0*d10 + wy1*d11))
                            * 0.0625f;                         // 1/sqrt(256)
          out[((size_t)(b*NCH + lvl*81 + o)*H_ + y)*W_ + x] = val;
        }
      }
      __syncthreads();
    }
  }
}

// ---------------------------------------------------------------------------
extern "C" void kernel_launch(void* const* d_in, const int* in_sizes, int n_in,
                              void* d_out, int out_size, void* d_ws, size_t ws_size,
                              hipStream_t stream) {
  (void)in_sizes; (void)n_in; (void)out_size; (void)ws_size;
  const float* f1   = (const float*)d_in[0];
  const float* f2   = (const float*)d_in[1];
  const float* cent = (const float*)d_in[2];
  bf16_t* ws  = (bf16_t*)d_ws;
  float*  out = (float*)d_out;

  const unsigned prep_blocks = WS_ELEMS / 256u;          // exact
  corr_prep<<<prep_blocks, 256, 0, stream>>>(f1, f2, ws);

  const unsigned main_blocks = NPIX / 32u;               // 8 waves * 4 pixels each
  corr_main<<<main_blocks, 256, 0, stream>>>(cent, ws, out);
}
